// MultiHeadAttention_70420283785685
// MI455X (gfx1250) — compile-verified, matrix operand order checked by reading, not run on hardware
//
#include <hip/hip_runtime.h>
#include <hip/hip_bf16.h>

// ---------------------------------------------------------------------------
// MI455X (gfx1250) GQA attention block: QKV proj + RoPE + causal flash attn +
// out proj.  Matmuls via v_wmma_f32_16x16x32_bf16 (wave32); tile staging via
// the Tensor Data Mover (tensor_load_to_lds + s_wait_tensorcnt) when the
// builtin is available, cooperative vector copies otherwise.
//   B=2, S=2048, H=2048, NQ=16, NKV=4, HD=128, theta=10000
// ---------------------------------------------------------------------------

#define B_DIM   2
#define S_LEN   2048
#define H_DIM   2048
#define NQ_H    16
#define NKV_H   4
#define HD_DIM  128
#define M_ROWS  (B_DIM * S_LEN)          // 4096 flattened (b,s) rows

typedef __attribute__((ext_vector_type(16))) __bf16 v16bf;
typedef __attribute__((ext_vector_type(8)))  float  v8f;

#if defined(__AMDGCN__) && __has_builtin(__builtin_amdgcn_tensor_load_to_lds) && \
    __has_builtin(__builtin_amdgcn_s_wait_tensorcnt)
#define HAVE_TDM 1
#endif

#ifdef HAVE_TDM
typedef __attribute__((ext_vector_type(4))) unsigned int v4ui;
typedef __attribute__((ext_vector_type(8))) int          v8si;
typedef __attribute__((ext_vector_type(4))) int          v4si;

// Build a 2D tensor-DMA descriptor (D#) and issue TENSOR_LOAD_TO_LDS.
// gptr      : global address of tile start (bf16 elements)
// lds_off   : LDS byte address of destination (low 32 bits of generic ptr)
// tile_d0/1 : tile size, elements (d0 = contiguous dim)
// tens_d0/1 : tensor extents for OOB checking (elements)
// stride0   : elements between consecutive dim-1 lines
__device__ __forceinline__ void tdm_load_2d(const void* gptr, unsigned lds_off,
                                            unsigned tile_d0, unsigned tile_d1,
                                            unsigned tens_d0, unsigned tens_d1,
                                            unsigned stride0) {
    unsigned long long ga = (unsigned long long)(size_t)gptr;
    v4ui g0;
    g0[0] = 1u;                                            // count=1, user D#
    g0[1] = lds_off;                                       // lds_addr (bytes)
    g0[2] = (unsigned)(ga & 0xFFFFFFFFu);                  // global_addr[31:0]
    g0[3] = (unsigned)((ga >> 32) & 0x01FFFFFFu)           // global_addr[56:32]
          | (2u << 30);                                    // type = 2 (image)
    v8si g1;
    g1[0] = (int)(1u << 16);                               // data_size=1 (2B)
    g1[1] = (int)((tens_d0 & 0xFFFFu) << 16);              // tensor_dim0 lo16
    g1[2] = (int)((tens_d0 >> 16) | ((tens_d1 & 0xFFFFu) << 16));
    g1[3] = (int)((tens_d1 >> 16) | (tile_d0 << 16));      // tile_dim0
    g1[4] = (int)(tile_d1);                                // tile_dim1 (dim2=0)
    g1[5] = (int)stride0;                                  // dim0 stride[31:0]
    g1[6] = 0;
    g1[7] = 0;
    v4si z4 = (v4si){0, 0, 0, 0};
    v8si z8 = (v8si){0, 0, 0, 0, 0, 0, 0, 0};
    // amdgpu-toolchain (clang-23) 6-arg form: (g0, g1, g2, g3, g4, cpol)
    __builtin_amdgcn_tensor_load_to_lds(g0, g1, z4, z4, z8, 0);
}
#endif

union FragU { v16bf v; uint4 u[2]; };

__device__ __forceinline__ unsigned short f2bf_u16(float f) {
    unsigned u = __float_as_uint(f);
    unsigned r = u + 0x7FFFu + ((u >> 16) & 1u);   // round-to-nearest-even
    return (unsigned short)(r >> 16);
}

// Load a 16x32 bf16 A/B fragment from LDS (row-major, p points at (row, kb)):
// VGPR0-3 = K[kb..kb+7], VGPR4-7 = K[kb+16..kb+23].
__device__ __forceinline__ v16bf ld_frag16(const unsigned short* p) {
    FragU f;
    f.u[0] = *(const uint4*)(p);
    f.u[1] = *(const uint4*)(p + 16);
    return f.v;
}

__device__ __forceinline__ v8f wmma_bf16(v16bf a, v16bf b, v8f c) {
    return __builtin_amdgcn_wmma_f32_16x16x32_bf16(
        /*neg_a=*/false, a, /*neg_b=*/false, b,
        /*c_mod=*/(short)0, c, /*reuse_a=*/false, /*reuse_b=*/false);
}

// ---------------------------------------------------------------------------
// 1) f32 -> bf16 conversion (weights + activations), grid-stride
// ---------------------------------------------------------------------------
__global__ void __launch_bounds__(256)
cvt_f32_bf16_kernel(const float* __restrict__ in, unsigned short* __restrict__ out, int n) {
    int i = blockIdx.x * blockDim.x + threadIdx.x;
    int stride = gridDim.x * blockDim.x;
    for (; i < n; i += stride) out[i] = f2bf_u16(in[i]);
}

// ---------------------------------------------------------------------------
// 2) GEMM  Y = X * W^T   (X: M x K bf16 row-major, W: N x K bf16 row-major)
//    Block tile 128(M) x 128(N), 8 waves, TDM double-buffered K chunks of 32.
//    mode 0: store bf16 to [b, head, s, d]
//    mode 1: same + fused RoPE (theta=10000, HD=128)
//    mode 2: store f32 to [row, nBase+col]  (final output projection)
// ---------------------------------------------------------------------------
__global__ void __launch_bounds__(256)
gemm_bf16_kernel(const unsigned short* __restrict__ X,
                 const unsigned short* __restrict__ W,
                 unsigned short* __restrict__ outB,
                 float* __restrict__ outF,
                 int K, int NH, int mode)
{
    __shared__ __align__(16) unsigned short lds_x[2][128 * 32];
    __shared__ __align__(16) unsigned short lds_w[2][128 * 32];

    const int t     = threadIdx.x;
    const int wid   = t >> 5;
    const int lane  = t & 31;
    const int r16   = lane & 15;
    const int kb    = (lane >> 4) << 3;   // K sub-offset for fragments
    const int hi8   = kb;                 // row offset for C/D layout
    const int mBase = blockIdx.x * 128;
    const int nBase = blockIdx.y * 128;

    v8f acc[8];
#pragma unroll
    for (int i = 0; i < 8; ++i) acc[i] = (v8f){0,0,0,0,0,0,0,0};

    const int nK = K >> 5;

#ifdef HAVE_TDM
    if (wid == 0) {   // prologue: DMA chunk 0 into buffer 0
        tdm_load_2d(&X[(size_t)mBase * K], (unsigned)(size_t)&lds_x[0][0],
                    32, 128, (unsigned)K, 1u << 20, (unsigned)K);
        tdm_load_2d(&W[(size_t)nBase * K], (unsigned)(size_t)&lds_w[0][0],
                    32, 128, (unsigned)K, 1u << 20, (unsigned)K);
    }
#endif

    for (int kc = 0; kc < nK; ++kc) {
        const int buf = kc & 1;
#ifdef HAVE_TDM
        if (wid == 0) {
            if (kc + 1 < nK) {   // DMA next chunk while this one is consumed
                tdm_load_2d(&X[(size_t)mBase * K + ((kc + 1) << 5)],
                            (unsigned)(size_t)&lds_x[buf ^ 1][0],
                            32, 128, (unsigned)K, 1u << 20, (unsigned)K);
                tdm_load_2d(&W[(size_t)nBase * K + ((kc + 1) << 5)],
                            (unsigned)(size_t)&lds_w[buf ^ 1][0],
                            32, 128, (unsigned)K, 1u << 20, (unsigned)K);
                __builtin_amdgcn_s_wait_tensorcnt(2);   // chunk kc landed
            } else {
                __builtin_amdgcn_s_wait_tensorcnt(0);
            }
        }
#else
        // cooperative staging fallback: 2 x uint4 per thread per tile
#pragma unroll
        for (int i = 0; i < 2; ++i) {
            int idx = t + (i << 8);      // 0..511
            int row = idx >> 2;          // 0..127
            int off = (idx & 3) << 3;    // 0,8,16,24
            *(uint4*)&lds_x[buf][row*32 + off] =
                *(const uint4*)&X[(size_t)(mBase + row)*K + (kc << 5) + off];
            *(uint4*)&lds_w[buf][row*32 + off] =
                *(const uint4*)&W[(size_t)(nBase + row)*K + (kc << 5) + off];
        }
#endif
        __syncthreads();

        v16bf aF = ld_frag16(&lds_x[buf][(wid*16 + r16)*32 + kb]);
        v16bf bF = ld_frag16(&lds_w[buf][(r16)*32 + kb]);
#pragma unroll
        for (int nt = 0; nt < 8; ++nt) {      // pipeline B-frag loads 1 ahead
            v16bf bN = bF;
            if (nt < 7) bN = ld_frag16(&lds_w[buf][((nt + 1)*16 + r16)*32 + kb]);
            acc[nt] = wmma_bf16(aF, bF, acc[nt]);
            bF = bN;
        }
        __syncthreads();
    }

    // ------------- epilogue -------------
    if (mode == 1) {
        // RoPE: d and d+64 live in acc[nt] and acc[nt+4], same lane/vgpr.
#pragma unroll
        for (int r = 0; r < 8; ++r) {
            int grow = mBase + wid*16 + r + hi8;
            int pos  = grow & (S_LEN - 1);         // position id == s
#pragma unroll
            for (int nt = 0; nt < 4; ++nt) {
                int   d   = nt*16 + r16;
                float inv = __powf(10000.0f, -(float)d * (1.0f / 64.0f));
                float ang = (float)pos * inv;
                float sn, cs;
                __sincosf(ang, &sn, &cs);
                float a  = acc[nt][r];
                float b2 = acc[nt + 4][r];
                acc[nt][r]     = a  * cs - b2 * sn;
                acc[nt + 4][r] = b2 * cs + a  * sn;
            }
        }
    }

    if (mode == 2) {
#pragma unroll
        for (int nt = 0; nt < 8; ++nt)
#pragma unroll
            for (int r = 0; r < 8; ++r) {
                int grow = mBase + wid*16 + r + hi8;
                outF[(size_t)grow * H_DIM + nBase + nt*16 + r16] = acc[nt][r];
            }
    } else {
        const int head = blockIdx.y;
#pragma unroll
        for (int nt = 0; nt < 8; ++nt)
#pragma unroll
            for (int r = 0; r < 8; ++r) {
                int grow = mBase + wid*16 + r + hi8;
                int bb   = grow >> 11;             // / S_LEN
                int ss   = grow & (S_LEN - 1);
                size_t addr = (((size_t)bb * NH + head) * S_LEN + ss) * HD_DIM
                            + nt*16 + r16;
                outB[addr] = f2bf_u16(acc[nt][r]);
            }
    }
}

// ---------------------------------------------------------------------------
// 3) Causal flash attention.  grid = (S/128, NQ, B), 8 waves/block.
//    Q tile + K chunks staged by TDM; V chunk transposed by the waves.
// ---------------------------------------------------------------------------
__global__ void __launch_bounds__(256)
flash_attn_kernel(const unsigned short* __restrict__ Q,
                  const unsigned short* __restrict__ Kb,
                  const unsigned short* __restrict__ Vb,
                  unsigned short* __restrict__ O)
{
    __shared__ __align__(16) unsigned short q_s[128 * 128];   // Q tile
    __shared__ __align__(16) unsigned short kt_s[32 * 128];   // K chunk [kv][d]
    __shared__ __align__(16) unsigned short vt_s[128 * 32];   // V chunk transposed [d][kv]
    __shared__ __align__(16) unsigned short p_s[8][16 * 32];  // per-wave P

    const int t    = threadIdx.x;
    const int wid  = t >> 5;
    const int lane = t & 31;
    const int r16  = lane & 15;
    const int kb   = (lane >> 4) << 3;
    const int hi8  = kb;

    const int qBase = blockIdx.x * 128;
    const int h     = blockIdx.y;
    const int b     = blockIdx.z;
    const int kvh   = h >> 2;                       // NQ/NKV = 4
    const int qw    = qBase + wid * 16;             // this wave's first q row

    const unsigned short* qhead = Q  + (((size_t)b * NQ_H  + h  ) * S_LEN) * HD_DIM;
    const unsigned short* khead = Kb + (((size_t)b * NKV_H + kvh) * S_LEN) * HD_DIM;
    const unsigned short* vhead = Vb + (((size_t)b * NKV_H + kvh) * S_LEN) * HD_DIM;

#ifdef HAVE_TDM
    if (wid == 0) {   // DMA the whole 128x128 Q tile
        tdm_load_2d(&qhead[(size_t)qBase * HD_DIM], (unsigned)(size_t)&q_s[0],
                    HD_DIM, 128, HD_DIM, 1u << 20, HD_DIM);
    }
#else
#pragma unroll
    for (int i = 0; i < 8; ++i) {
        int idx = t + (i << 8);          // 0..2047
        int row = idx >> 4;              // 16 uint4 per row
        int off = (idx & 15) << 3;
        *(uint4*)&q_s[row*HD_DIM + off] =
            *(const uint4*)&qhead[(size_t)(qBase + row)*HD_DIM + off];
    }
#endif

    v8f accO[8];
#pragma unroll
    for (int i = 0; i < 8; ++i) accO[i] = (v8f){0,0,0,0,0,0,0,0};
    float mrow[8], lrow[8];
#pragma unroll
    for (int r = 0; r < 8; ++r) { mrow[r] = -1e30f; lrow[r] = 0.0f; }

    const int   nChunks = blockIdx.x * 4 + 4;             // causal extent
    const float scale   = 0.08838834764831845f;           // 1/sqrt(128)

    for (int j = 0; j < nChunks; ++j) {
        __syncthreads();
#ifdef HAVE_TDM
        if (wid == 0) {   // DMA K chunk [32 x 128]; wait covers Q tile too
            tdm_load_2d(&khead[(size_t)(j*32) * HD_DIM],
                        (unsigned)(size_t)&kt_s[0],
                        HD_DIM, 32, HD_DIM, 1u << 20, HD_DIM);
            __builtin_amdgcn_s_wait_tensorcnt(0);
        }
        // V chunk transposed by the waves (TDM cannot transpose)
#pragma unroll
        for (int i = 0; i < 2; ++i) {
            int idx = t + (i << 8);
            int row = idx >> 4;
            int off = (idx & 15) << 3;
            uint4 vv = *(const uint4*)&vhead[(size_t)(j*32 + row)*HD_DIM + off];
            const unsigned short* pv = (const unsigned short*)&vv;
#pragma unroll
            for (int e = 0; e < 8; ++e) vt_s[(off + e)*32 + row] = pv[e];
        }
#else
#pragma unroll
        for (int i = 0; i < 2; ++i) {
            int idx = t + (i << 8);      // 0..511
            int row = idx >> 4;          // kv row 0..31
            int off = (idx & 15) << 3;   // d offset
            *(uint4*)&kt_s[row*HD_DIM + off] =
                *(const uint4*)&khead[(size_t)(j*32 + row)*HD_DIM + off];
            uint4 vv = *(const uint4*)&vhead[(size_t)(j*32 + row)*HD_DIM + off];
            const unsigned short* pv = (const unsigned short*)&vv;
#pragma unroll
            for (int e = 0; e < 8; ++e) vt_s[(off + e)*32 + row] = pv[e];
        }
#endif
        __syncthreads();

        if (j * 32 <= qw + 15) {          // wave-uniform causal skip
            // ---- S = Q K^T : 16(q) x 32(kv), contraction over HD=128 ----
            v8f s0 = (v8f){0,0,0,0,0,0,0,0};
            v8f s1 = (v8f){0,0,0,0,0,0,0,0};
#pragma unroll
            for (int kk = 0; kk < 4; ++kk) {
                v16bf aF = ld_frag16(&q_s[(wid*16 + r16)*HD_DIM + kk*32 + kb]);
                v16bf b0 = ld_frag16(&kt_s[(r16     )*HD_DIM + kk*32 + kb]);
                v16bf b1 = ld_frag16(&kt_s[(16 + r16)*HD_DIM + kk*32 + kb]);
                s0 = wmma_bf16(aF, b0, s0);
                s1 = wmma_bf16(aF, b1, s1);
            }
            // ---- online softmax (f32) ----
#pragma unroll
            for (int r = 0; r < 8; ++r) {
                int   qpos = qw + r + hi8;
                int   kv0  = j*32 + r16;
                float a0 = (kv0      <= qpos) ? s0[r] * scale : -1e30f;
                float a1 = (kv0 + 16 <= qpos) ? s1[r] * scale : -1e30f;
                float rm = fmaxf(a0, a1);
#pragma unroll
                for (int mm = 1; mm <= 8; mm <<= 1)
                    rm = fmaxf(rm, __shfl_xor(rm, mm, 32));
                float mnew = fmaxf(mrow[r], rm);
                float corr = __expf(mrow[r] - mnew);
                float p0   = __expf(a0 - mnew);
                float p1   = __expf(a1 - mnew);
                float ps   = p0 + p1;
#pragma unroll
                for (int mm = 1; mm <= 8; mm <<= 1)
                    ps += __shfl_xor(ps, mm, 32);
                mrow[r] = mnew;
                lrow[r] = lrow[r] * corr + ps;
#pragma unroll
                for (int nt = 0; nt < 8; ++nt) accO[nt][r] *= corr;
                // scatter P to LDS (C-layout -> A-fragment layout round trip)
                p_s[wid][(r + hi8)*32 + r16     ] = f2bf_u16(p0);
                p_s[wid][(r + hi8)*32 + 16 + r16] = f2bf_u16(p1);
            }
            // ---- O += P V : 16(q) x 128(d), contraction over 32 kv ----
            v16bf aP = ld_frag16(&p_s[wid][r16*32 + kb]);
            v16bf bV = ld_frag16(&vt_s[(r16)*32 + kb]);
#pragma unroll
            for (int nt = 0; nt < 8; ++nt) {
                v16bf bN = bV;
                if (nt < 7) bN = ld_frag16(&vt_s[((nt + 1)*16 + r16)*32 + kb]);
                accO[nt] = wmma_bf16(aP, bV, accO[nt]);
                bV = bN;
            }
        }
    }

    // normalize + store O as bf16 row-major [b*S+s, h*128+d] for out-proj GEMM
#pragma unroll
    for (int nt = 0; nt < 8; ++nt)
#pragma unroll
        for (int r = 0; r < 8; ++r) {
            int   qpos = qw + r + hi8;
            float o    = accO[nt][r] / lrow[r];
            O[((size_t)(b * S_LEN + qpos)) * H_DIM + h * HD_DIM + nt*16 + r16] =
                f2bf_u16(o);
        }
}

// ---------------------------------------------------------------------------
// host-side launch
// ---------------------------------------------------------------------------
extern "C" void kernel_launch(void* const* d_in, const int* in_sizes, int n_in,
                              void* d_out, int out_size, void* d_ws, size_t ws_size,
                              hipStream_t stream) {
    (void)in_sizes; (void)n_in; (void)out_size; (void)ws_size;

    const float* hs = (const float*)d_in[0];
    // d_in[1] = position_ids (arange; RoPE uses s directly)
    const float* wq = (const float*)d_in[2];
    const float* wk = (const float*)d_in[3];
    const float* wv = (const float*)d_in[4];
    const float* wo = (const float*)d_in[5];
    float* out = (float*)d_out;

    size_t off = 0;
    auto alloc16 = [&](size_t elems) {
        unsigned short* p = (unsigned short*)((char*)d_ws + off);
        off = (off + elems * 2 + 255) & ~(size_t)255;
        return p;
    };
    unsigned short* hs_bf = alloc16((size_t)M_ROWS * H_DIM);
    unsigned short* wq_bf = alloc16((size_t)NQ_H  * HD_DIM * H_DIM);
    unsigned short* wk_bf = alloc16((size_t)NKV_H * HD_DIM * H_DIM);
    unsigned short* wv_bf = alloc16((size_t)NKV_H * HD_DIM * H_DIM);
    unsigned short* wo_bf = alloc16((size_t)H_DIM * H_DIM);
    unsigned short* q_buf = alloc16((size_t)B_DIM * NQ_H  * S_LEN * HD_DIM);
    unsigned short* k_buf = alloc16((size_t)B_DIM * NKV_H * S_LEN * HD_DIM);
    unsigned short* v_buf = alloc16((size_t)B_DIM * NKV_H * S_LEN * HD_DIM);
    unsigned short* o_buf = alloc16((size_t)M_ROWS * H_DIM);

    // 1) precision conversion
    cvt_f32_bf16_kernel<<<2048, 256, 0, stream>>>(hs, hs_bf, M_ROWS * H_DIM);
    cvt_f32_bf16_kernel<<<1024, 256, 0, stream>>>(wq, wq_bf, NQ_H  * HD_DIM * H_DIM);
    cvt_f32_bf16_kernel<<< 512, 256, 0, stream>>>(wk, wk_bf, NKV_H * HD_DIM * H_DIM);
    cvt_f32_bf16_kernel<<< 512, 256, 0, stream>>>(wv, wv_bf, NKV_H * HD_DIM * H_DIM);
    cvt_f32_bf16_kernel<<<1024, 256, 0, stream>>>(wo, wo_bf, H_DIM * H_DIM);

    // 2) QKV projections (RoPE fused into Q and K)
    gemm_bf16_kernel<<<dim3(32, NQ_H),  256, 0, stream>>>(hs_bf, wq_bf, q_buf, nullptr, H_DIM, NQ_H,  1);
    gemm_bf16_kernel<<<dim3(32, NKV_H), 256, 0, stream>>>(hs_bf, wk_bf, k_buf, nullptr, H_DIM, NKV_H, 1);
    gemm_bf16_kernel<<<dim3(32, NKV_H), 256, 0, stream>>>(hs_bf, wv_bf, v_buf, nullptr, H_DIM, NKV_H, 0);

    // 3) causal flash attention
    flash_attn_kernel<<<dim3(S_LEN / 128, NQ_H, B_DIM), 256, 0, stream>>>(q_buf, k_buf, v_buf, o_buf);

    // 4) output projection -> f32
    gemm_bf16_kernel<<<dim3(32, H_DIM / 128), 256, 0, stream>>>(o_buf, wo_bf, nullptr, out, H_DIM, 0, 2);
}
